// FDGNN_83382495084586
// MI455X (gfx1250) — matrix operand order repaired and checked by reading, not compiled, not run on hardware
//
#include <hip/hip_runtime.h>
#include <math.h>

// ---------------------------------------------------------------------------
// STAEformer-style forward for MI455X (gfx1250, wave32, WMMA).
// GEMMs: v_wmma_f32_16x16x32_f16, B panels staged in LDS (WGP has 320 KB),
// double-buffered ds_load_b128 B fragments, 32-row tiles to halve LDS traffic
// per WMMA. Attention/LN on VALU (hd=44 / L=307 are not WMMA shapes).
// ---------------------------------------------------------------------------

typedef __attribute__((ext_vector_type(16))) _Float16 v16h;
typedef __attribute__((ext_vector_type(8)))  float    v8f;

constexpr int kB   = 32;
constexpr int kT   = 12;
constexpr int kN   = 307;
constexpr int kD   = 176;
constexpr int kH   = 4;
constexpr int kHD  = 44;
constexpr int kFF  = 256;
constexpr int kTok = kB * kT * kN;            // 117888, multiple of 128
constexpr float kScale = 0.150755672288882f;  // 1/sqrt(44)

// Per-layer f16 weight arena (halves): Wq,Wk,Wv,Wo (176x192 each, transposed,
// K padded 176->192), W1t (256x192), W2t (176x256).
constexpr int OFF_WQ = 0;
constexpr int OFF_WK = 176 * 192;
constexpr int OFF_WV = 2 * 176 * 192;
constexpr int OFF_WO = 3 * 176 * 192;
constexpr int OFF_W1 = 4 * 176 * 192;
constexpr int OFF_W2 = 4 * 176 * 192 + 256 * 192;
constexpr int PER_LAYER = 4 * 176 * 192 + 256 * 192 + 176 * 256;  // 229376

// ---------------------------------------------------------------------------
// Weight prep: W (K x Nc, f32, row-major) -> Wt (Nc x Kpad, f16), zero-padded
// so every B fragment is a contiguous, 32B-aligned v16h run.
// ---------------------------------------------------------------------------
__global__ void convert_wt(const float* __restrict__ W, _Float16* __restrict__ Wt,
                           int K, int Nc, int Kpad) {
  int idx = blockIdx.x * blockDim.x + threadIdx.x;
  if (idx >= Nc * Kpad) return;
  int col = idx / Kpad;
  int kk  = idx % Kpad;
  float v = (kk < K) ? W[(size_t)kk * Nc + col] : 0.f;
  Wt[idx] = (_Float16)v;
}

// ---------------------------------------------------------------------------
// WMMA GEMM: Y[M,Nc] = act(X[M,K] @ W[K,Nc] + bias).
//  - Block (8 waves) stages the full f16 weight panel (Nc x Kpad) in LDS once.
//  - Wave owns ROWS*16 rows x all NTILES column tiles. Each B fragment from
//    LDS feeds ROWS WMMAs (halves LDS bandwidth per matrix op for ROWS=2);
//    each A fragment is reused NTILES times.
//  - B fragments are double-buffered so a ds_load pair is always in flight
//    under the current WMMA (wait becomes dscnt<=2 instead of 0).
//  - A loads are float4-vectorized; K-tail correctness comes from the zeroed
//    B padding (A tail garbage multiplies by 0). <=64B row overrun stays
//    inside the workspace (X is never the last ws region).
// ---------------------------------------------------------------------------
template <int NTILES, int ROWS>
__global__ void __launch_bounds__(256)
gemm_wmma(const float* __restrict__ X, const _Float16* __restrict__ Wt,
          const float* __restrict__ bias, float* __restrict__ Y,
          int M, int K, int Kpad, int relu) {
  extern __shared__ _Float16 sW[];  // Nc x Kpad halves
  const int Nc   = NTILES * 16;
  const int wave = (blockIdx.x * blockDim.x + threadIdx.x) >> 5;
  const int lane = threadIdx.x & 31;
  const int rowBase = wave * (16 * ROWS);

  // Cooperative weight-panel stage (uint4 = 8 halves per transfer).
  {
    const int chunks = (Nc * Kpad) / 8;
    const uint4* src = (const uint4*)Wt;
    uint4* dst = (uint4*)sW;
    for (int i = threadIdx.x; i < chunks; i += 256) dst[i] = src[i];
  }
  __syncthreads();
  if (rowBase >= M) return;  // wave-uniform (EXEC all-ones for WMMA)

  const int hiHalf = (lane >= 16) ? 1 : 0;
  const int l15    = lane & 15;

  v8f acc[ROWS][NTILES];
#pragma unroll
  for (int r = 0; r < ROWS; r++)
#pragma unroll
    for (int i = 0; i < NTILES; i++) acc[r][i] = {};

  const float* xrow[ROWS];
#pragma unroll
  for (int r = 0; r < ROWS; r++)
    xrow[r] = X + (size_t)(rowBase + r * 16 + l15) * K;

  const _Float16* sWlane = sW + l15 * (size_t)Kpad;

  for (int k = 0; k < K; k += 32) {
#pragma unroll
    for (int r = 0; r < ROWS; r++)
      __builtin_prefetch(xrow[r] + k + 32, 0, 1);  // global_prefetch_b8

    // A fragments: lanes 0-15 row M=l15 hold
    // K = k + {0..15}; lanes 16-31 hold K = k + {16..31} (matches B).
    const int kbA = k + hiHalf * 16;
    v16h a[ROWS];
#pragma unroll
    for (int r = 0; r < ROWS; r++) {
      const float4* xp = (const float4*)(xrow[r] + kbA);       // 16B-aligned
      const float4* xq = (const float4*)(xrow[r] + kbA + 8);   // 16B-aligned
      float4 r0 = xp[0], r1 = xp[1], r2 = xq[0], r3 = xq[1];
      a[r][0]  = (_Float16)r0.x; a[r][1]  = (_Float16)r0.y;
      a[r][2]  = (_Float16)r0.z; a[r][3]  = (_Float16)r0.w;
      a[r][4]  = (_Float16)r1.x; a[r][5]  = (_Float16)r1.y;
      a[r][6]  = (_Float16)r1.z; a[r][7]  = (_Float16)r1.w;
      a[r][8]  = (_Float16)r2.x; a[r][9]  = (_Float16)r2.y;
      a[r][10] = (_Float16)r2.z; a[r][11] = (_Float16)r2.w;
      a[r][12] = (_Float16)r3.x; a[r][13] = (_Float16)r3.y;
      a[r][14] = (_Float16)r3.z; a[r][15] = (_Float16)r3.w;
    }

    // B fragments from LDS: lane col = nt*16+l15; lanes 0-15 K=k..k+15,
    // lanes 16-31 K=k+16..k+31; 32B-contiguous per lane. Double-buffered.
    const int kbB = k + hiHalf * 16;
    v16h bf[2];
    bf[0] = *(const v16h*)(sWlane + kbB);
#pragma unroll
    for (int nt = 0; nt < NTILES; nt++) {
      if (nt + 1 < NTILES)
        bf[(nt + 1) & 1] = *(const v16h*)(sWlane + (size_t)(nt + 1) * 16 * Kpad + kbB);
#pragma unroll
      for (int r = 0; r < ROWS; r++)
        acc[r][nt] = __builtin_amdgcn_wmma_f32_16x16x32_f16(
            false, a[r], false, bf[nt & 1], (short)0, acc[r][nt], false, false);
    }
  }

  // Epilogue per 16x16 C/D layout: VGPR j -> row tileBase + j + 8*hiHalf.
#pragma unroll
  for (int r = 0; r < ROWS; r++) {
    const int rowOff = rowBase + r * 16 + hiHalf * 8;
#pragma unroll
    for (int nt = 0; nt < NTILES; nt++) {
      int col = nt * 16 + l15;
      float bv = bias[col];
#pragma unroll
      for (int j = 0; j < 8; j++) {
        float y = acc[r][nt][j] + bv;
        if (relu) y = fmaxf(y, 0.f);
        Y[(size_t)(rowOff + j) * Nc + col] = y;
      }
    }
  }
}

// ---------------------------------------------------------------------------
// Embedding: h = concat(x@W_in+b_in, tod_emb, dow_emb, node_emb, adp_emb).
// ---------------------------------------------------------------------------
__global__ void embed_kernel(const float* __restrict__ x,
                             const float* __restrict__ W_in, const float* __restrict__ b_in,
                             const float* __restrict__ tod_emb, const float* __restrict__ dow_emb,
                             const float* __restrict__ node_emb, const float* __restrict__ adp_emb,
                             float* __restrict__ h) {
  int idx = blockIdx.x * blockDim.x + threadIdx.x;
  if (idx >= kTok * kD) return;
  int m = idx / kD, d = idx % kD;
  int n = m % kN;
  int t = (m / kN) % kT;
  const float* xr = x + (size_t)m * 3;
  float val;
  if (d < 24) {
    val = b_in[d] + xr[0] * W_in[d] + xr[1] * W_in[24 + d] + xr[2] * W_in[48 + d];
  } else if (d < 48) {
    int ti = (int)(xr[1] * 288.f);
    ti = min(max(ti, 0), 287);
    val = tod_emb[ti * 24 + (d - 24)];
  } else if (d < 72) {
    int di = (int)xr[2];
    di = min(max(di, 0), 6);
    val = dow_emb[di * 24 + (d - 48)];
  } else if (d < 96) {
    val = node_emb[n * 24 + (d - 72)];
  } else {
    val = adp_emb[(size_t)(t * kN + n) * 80 + (d - 96)];
  }
  h[idx] = val;
}

// ---------------------------------------------------------------------------
// Temporal attention (L = 12): one thread per (b,n,head,t) query; softmax in
// registers; output written in-place into q.
// ---------------------------------------------------------------------------
__global__ void attn_temporal(float* __restrict__ q, const float* __restrict__ k,
                              const float* __restrict__ v) {
  int idx = blockIdx.x * blockDim.x + threadIdx.x;
  if (idx >= kB * kN * kH * kT) return;
  int n = idx % kN;
  int rest = idx / kN;
  int hh = rest % kH;
  rest /= kH;
  int t = rest % kT;
  int b = rest / kT;

  const size_t kstride = (size_t)kN * kD;
  const size_t base0   = ((size_t)(b * kT) * kN + n) * kD + hh * kHD;
  const size_t qbase   = base0 + (size_t)t * kstride;

  float qv[kHD];
#pragma unroll
  for (int d = 0; d < kHD; d++) qv[d] = q[qbase + d];

  float s[kT];
  float mmax = -1e30f;
  for (int t2 = 0; t2 < kT; t2++) {
    const float* kp = k + base0 + (size_t)t2 * kstride;
    float acc = 0.f;
#pragma unroll
    for (int d = 0; d < kHD; d++) acc += qv[d] * kp[d];
    acc *= kScale;
    s[t2] = acc;
    mmax = fmaxf(mmax, acc);
  }
  float l = 0.f;
  for (int t2 = 0; t2 < kT; t2++) { s[t2] = __expf(s[t2] - mmax); l += s[t2]; }
  float inv = 1.f / l;

  float o[kHD];
#pragma unroll
  for (int d = 0; d < kHD; d++) o[d] = 0.f;
  for (int t2 = 0; t2 < kT; t2++) {
    const float* vp = v + base0 + (size_t)t2 * kstride;
    float w = s[t2];
#pragma unroll
    for (int d = 0; d < kHD; d++) o[d] += w * vp[d];
  }
#pragma unroll
  for (int d = 0; d < kHD; d++) q[qbase + d] = o[d] * inv;
}

// ---------------------------------------------------------------------------
// Spatial attention (L = 307): one block per (b,t,head). K,V staged in dynamic
// LDS (2 * 307*44*4 = 108 KB of the WGP's 320 KB); online softmax per query.
// ---------------------------------------------------------------------------
__global__ void __launch_bounds__(256)
attn_spatial(float* __restrict__ q, const float* __restrict__ k,
             const float* __restrict__ v) {
  extern __shared__ float smem[];
  float* sK = smem;
  float* sV = smem + kN * kHD;

  int blk = blockIdx.x;            // (b*kT + t)*kH + hh
  int hh  = blk % kH;
  int bt  = blk / kH;
  const size_t tokbase = (size_t)bt * kN * kD + hh * kHD;

  for (int i = threadIdx.x; i < kN * kHD; i += blockDim.x) {
    int nn = i / kHD, d = i % kHD;
    size_t g = tokbase + (size_t)nn * kD + d;
    sK[i] = k[g];
    sV[i] = v[g];
  }
  __syncthreads();

  for (int nq = threadIdx.x; nq < kN; nq += blockDim.x) {
    const size_t qbase = tokbase + (size_t)nq * kD;
    float qv[kHD];
#pragma unroll
    for (int d = 0; d < kHD; d++) qv[d] = q[qbase + d];
    float m = -1e30f, l = 0.f;
    float o[kHD];
#pragma unroll
    for (int d = 0; d < kHD; d++) o[d] = 0.f;
    for (int j = 0; j < kN; j++) {
      const float* kp = sK + j * kHD;
      float sc = 0.f;
#pragma unroll
      for (int d = 0; d < kHD; d++) sc += qv[d] * kp[d];
      sc *= kScale;
      float mn   = fmaxf(m, sc);
      float corr = __expf(m - mn);
      float w    = __expf(sc - mn);
      l = l * corr + w;
      const float* vp = sV + j * kHD;
#pragma unroll
      for (int d = 0; d < kHD; d++) o[d] = o[d] * corr + w * vp[d];
      m = mn;
    }
    float inv = 1.f / l;
#pragma unroll
    for (int d = 0; d < kHD; d++) q[qbase + d] = o[d] * inv;
  }
}

// ---------------------------------------------------------------------------
// Residual + LayerNorm over D=176: one wave32 per token, shfl_xor reductions.
// h = LN(h + r) * g + beta, in place.
// ---------------------------------------------------------------------------
__global__ void __launch_bounds__(256)
ln_kernel(float* __restrict__ h, const float* __restrict__ r,
          const float* __restrict__ g, const float* __restrict__ beta) {
  int wave = (blockIdx.x * blockDim.x + threadIdx.x) >> 5;
  int lane = threadIdx.x & 31;
  if (wave >= kTok) return;
  const size_t base = (size_t)wave * kD;

  float xv[6];
  float s = 0.f;
#pragma unroll
  for (int j = 0; j < 6; j++) {
    int d = lane + j * 32;
    float val = 0.f;
    if (d < kD) val = h[base + d] + r[base + d];
    xv[j] = val;
    s += val;
  }
#pragma unroll
  for (int m = 16; m >= 1; m >>= 1) s += __shfl_xor(s, m, 32);
  float mu = s * (1.f / kD);

  float vs = 0.f;
#pragma unroll
  for (int j = 0; j < 6; j++) {
    int d = lane + j * 32;
    if (d < kD) { float c = xv[j] - mu; vs += c * c; }
  }
#pragma unroll
  for (int m = 16; m >= 1; m >>= 1) vs += __shfl_xor(vs, m, 32);
  float inv = rsqrtf(vs * (1.f / kD) + 1e-5f);

#pragma unroll
  for (int j = 0; j < 6; j++) {
    int d = lane + j * 32;
    if (d < kD) h[base + d] = (xv[j] - mu) * inv * g[d] + beta[d];
  }
}

// ---------------------------------------------------------------------------
// Output projection: out[b,to,n] = sum_{t,d} h[b,t,n,d] * Wout[t*176+d, to].
// 0.5 GFLOP total -> plain VALU.
// ---------------------------------------------------------------------------
__global__ void out_proj(const float* __restrict__ h, const float* __restrict__ Wout,
                         const float* __restrict__ bout, float* __restrict__ out) {
  int idx = blockIdx.x * blockDim.x + threadIdx.x;
  if (idx >= kB * kT * kN) return;
  int n  = idx % kN;
  int to = (idx / kN) % kT;
  int b  = idx / (kN * kT);
  float acc = bout[to];
  for (int t = 0; t < kT; t++) {
    const float* hp = h + ((size_t)(b * kT + t) * kN + n) * kD;
    const float* wp = Wout + (size_t)t * kD * 12 + to;
    for (int d = 0; d < kD; d++) acc += hp[d] * wp[(size_t)d * 12];
  }
  out[idx] = acc;
}

// ---------------------------------------------------------------------------
// Host orchestration.
// Workspace (floats): [h | q | k | v | wf16(halves)], FFN mid overlays k..v.
// Total ~335 MB, all regions fully written before any read.
// ---------------------------------------------------------------------------
extern "C" void kernel_launch(void* const* d_in, const int* in_sizes, int n_in,
                              void* d_out, int out_size, void* d_ws, size_t ws_size,
                              hipStream_t stream) {
  (void)in_sizes; (void)n_in; (void)out_size; (void)ws_size;
  const float* x        = (const float*)d_in[0];
  const float* W_in     = (const float*)d_in[1];
  const float* b_in     = (const float*)d_in[2];
  const float* tod_emb  = (const float*)d_in[3];
  const float* dow_emb  = (const float*)d_in[4];
  const float* node_emb = (const float*)d_in[5];
  const float* adp_emb  = (const float*)d_in[6];
  const float* Wout     = (const float*)d_in[7];
  const float* bout     = (const float*)d_in[8];

  float* ws = (float*)d_ws;
  const size_t MD = (size_t)kTok * kD;
  float* h  = ws;
  float* qb = ws + MD;
  float* kb = ws + 2 * MD;
  float* vb = ws + 3 * MD;
  float* fb = kb;  // FFN mid (Tok x 256) overlays k..v (Tok x 352)
  _Float16* wf16 = (_Float16*)(ws + 4 * MD);  // 32B-aligned (4*MD*4 % 32 == 0)

  // 1) Convert all weights to transposed, K-padded f16 once per call.
  for (int li = 0; li < 6; li++) {
    int p = li / 3, i = li % 3;
    int base = 9 + p * 16;  // inputs 9..24 = temporal params, 25..40 = spatial
    const float* Wq = (const float*)d_in[base + 0]  + (size_t)i * kD * kD;
    const float* Wk = (const float*)d_in[base + 2]  + (size_t)i * kD * kD;
    const float* Wv = (const float*)d_in[base + 4]  + (size_t)i * kD * kD;
    const float* Wo = (const float*)d_in[base + 6]  + (size_t)i * kD * kD;
    const float* W1 = (const float*)d_in[base + 8]  + (size_t)i * kD * kFF;
    const float* W2 = (const float*)d_in[base + 10] + (size_t)i * kFF * kD;
    _Float16* wf = wf16 + (size_t)li * PER_LAYER;
    convert_wt<<<(176 * 192 + 255) / 256, 256, 0, stream>>>(Wq, wf + OFF_WQ, 176, 176, 192);
    convert_wt<<<(176 * 192 + 255) / 256, 256, 0, stream>>>(Wk, wf + OFF_WK, 176, 176, 192);
    convert_wt<<<(176 * 192 + 255) / 256, 256, 0, stream>>>(Wv, wf + OFF_WV, 176, 176, 192);
    convert_wt<<<(176 * 192 + 255) / 256, 256, 0, stream>>>(Wo, wf + OFF_WO, 176, 176, 192);
    convert_wt<<<(256 * 192 + 255) / 256, 256, 0, stream>>>(W1, wf + OFF_W1, 176, 256, 192);
    convert_wt<<<(176 * 256 + 255) / 256, 256, 0, stream>>>(W2, wf + OFF_W2, 256, 176, 256);
  }

  // 2) Embedding front-end.
  embed_kernel<<<(kTok * kD + 255) / 256, 256, 0, stream>>>(
      x, W_in, b_in, tod_emb, dow_emb, node_emb, adp_emb, h);

  // Grid sizes: ROWS=2 GEMMs cover 32 rows/wave (3684 waves -> 461 blocks,
  // guarded tail); ROWS=1 FFN GEMM covers 16 rows/wave (921 blocks exact).
  const int gemmBlocks2 = (kTok / 32 + 7) / 8;  // 461
  const int gemmBlocks1 = (kTok / 16) / 8;      // 921
  const int lnBlocks    = kTok / 8;             // 8 wave-tokens per block
  const int lds176_192  = 176 * 192 * 2;        // 67.6 KB weight panel
  const int lds256_192  = 256 * 192 * 2;        // 98.3 KB
  const int lds176_256  = 176 * 256 * 2;        // 90.1 KB

  // 3) 3 temporal + 3 spatial self-attention layers.
  for (int li = 0; li < 6; li++) {
    int p = li / 3, i = li % 3;
    int base = 9 + p * 16;
    _Float16* wf = wf16 + (size_t)li * PER_LAYER;
    const float* bq  = (const float*)d_in[base + 1]  + (size_t)i * kD;
    const float* bk_ = (const float*)d_in[base + 3]  + (size_t)i * kD;
    const float* bv_ = (const float*)d_in[base + 5]  + (size_t)i * kD;
    const float* bo  = (const float*)d_in[base + 7]  + (size_t)i * kD;
    const float* b1  = (const float*)d_in[base + 9]  + (size_t)i * kFF;
    const float* b2  = (const float*)d_in[base + 11] + (size_t)i * kD;
    const float* g1  = (const float*)d_in[base + 12] + (size_t)i * kD;
    const float* B1  = (const float*)d_in[base + 13] + (size_t)i * kD;
    const float* g2  = (const float*)d_in[base + 14] + (size_t)i * kD;
    const float* B2  = (const float*)d_in[base + 15] + (size_t)i * kD;

    // Q,K,V projections (WMMA, LDS-staged weight panels, 32-row tiles).
    gemm_wmma<11, 2><<<gemmBlocks2, 256, lds176_192, stream>>>(h, wf + OFF_WQ, bq,  qb, kTok, 176, 192, 0);
    gemm_wmma<11, 2><<<gemmBlocks2, 256, lds176_192, stream>>>(h, wf + OFF_WK, bk_, kb, kTok, 176, 192, 0);
    gemm_wmma<11, 2><<<gemmBlocks2, 256, lds176_192, stream>>>(h, wf + OFF_WV, bv_, vb, kTok, 176, 192, 0);

    // Attention (writes in place into qb).
    if (p == 0) {
      attn_temporal<<<(kB * kN * kH * kT) / 256, 256, 0, stream>>>(qb, kb, vb);
    } else {
      attn_spatial<<<kB * kT * kH, 256, 2 * kN * kHD * sizeof(float), stream>>>(qb, kb, vb);
    }

    // Output projection -> kb; h = LN(h + kb).
    gemm_wmma<11, 2><<<gemmBlocks2, 256, lds176_192, stream>>>(qb, wf + OFF_WO, bo, kb, kTok, 176, 192, 0);
    ln_kernel<<<lnBlocks, 256, 0, stream>>>(h, kb, g1, B1);

    // FFN: relu(h@W1+b1) -> fb; fb@W2+b2 -> qb; h = LN(h + qb).
    gemm_wmma<16, 1><<<gemmBlocks1, 256, lds256_192, stream>>>(h,  wf + OFF_W1, b1, fb, kTok, 176, 192, 1);
    gemm_wmma<11, 2><<<gemmBlocks2, 256, lds176_256, stream>>>(fb, wf + OFF_W2, b2, qb, kTok, 256, 256, 0);
    ln_kernel<<<lnBlocks, 256, 0, stream>>>(h, qb, g2, B2);
  }

  // 4) Output projection -> (B, 12, N, 1).
  out_proj<<<(kTok + 255) / 256, 256, 0, stream>>>(h, Wout, bout, (float*)d_out);
}
